// DilatedAttention_54004918779914
// MI455X (gfx1250) — compile-verified, matrix operand order checked by reading, not run on hardware
//
#include <hip/hip_runtime.h>
#include <hip/hip_bf16.h>

// ---------------------------------------------------------------------------
// Dilated attention (R=2 on the sequence axis) for MI455X / gfx1250, wave32.
// All matmuls on v_wmma_f32_16x16x32_f16 (fp32 accumulate).
// V is stored transposed per head so every WMMA operand in the attention
// inner loop is two contiguous global_load_b128s.
// ---------------------------------------------------------------------------

typedef __attribute__((ext_vector_type(16))) _Float16     v16h;
typedef __attribute__((ext_vector_type(8)))  float        v8f;
typedef __attribute__((ext_vector_type(8)))  unsigned int v8u;

#define B_    2
#define S_    4096
#define HID_  1024
#define NH_   16
#define HD_   64
#define SD_   2048          // S_/R  (even sequence positions)
#define NEG_INF (-3.0e38f)

union HU { v16h h; v8u u; };

// K-offset of dword pair p (0..7) inside a 16-bit WMMA A/B operand, per the
// documented 16x32 layout: lanes 0-15 hold K {0..7,16..23}, lanes 16-31 add 8.
__device__ __forceinline__ int wmma_kbase(int p, int hi) {
    return ((p < 4) ? (2 * p) : (2 * p + 8)) + (hi ? 8 : 0);
}

// Gather a 16x32 f16 operand whose 32-element K runs are CONTIGUOUS in memory
// (compiler coalesces this into two global_load_b128 per operand).
// Element (r, k) lives at src[(r0+r)*ld + k0 + k].  Used for:
//   A = activations (row-major, K = feature dim)
//   B = W[o][h] weight tiles (column o of B == row o of W, contiguous in h)
//   B = K^T tiles (column kc of B == row kc of K, contiguous in d)
//   B = V tiles via the transposed Vt layout (column d == row d of Vt)
__device__ __forceinline__ v16h wmma_load_contig(const _Float16* src, int ld,
                                                 int r0, int k0) {
    int lane = threadIdx.x & 31;
    int hi   = lane >> 4;
    const _Float16* p = src + (size_t)(r0 + (lane & 15)) * ld + k0;
    HU c;
#pragma unroll
    for (int i = 0; i < 8; ++i)
        c.u[i] = *(const unsigned int*)(p + wmma_kbase(i, hi));
    return c.h;
}

__device__ __forceinline__ v8f wmma_f16(v16h a, v16h b, v8f c) {
    return __builtin_amdgcn_wmma_f32_16x16x32_f16(false, a, false, b,
                                                  (short)0, c, false, false);
}

// Row-wise reductions across the 16 lanes of a wave half (xor masks 1..8 keep
// the two halves — rows 0-7 vs 8-15 of the C/D layout — independent).
__device__ __forceinline__ float rowmax16(float v) {
#pragma unroll
    for (int m = 8; m >= 1; m >>= 1) v = fmaxf(v, __shfl_xor(v, m, 32));
    return v;
}
__device__ __forceinline__ float rowsum16(float v) {
#pragma unroll
    for (int m = 8; m >= 1; m >>= 1) v += __shfl_xor(v, m, 32);
    return v;
}

// ---------------------------------------------------------------------------
// Kernel 0: fp32 -> f16 conversion.  Even hidden rows only; 4 weight mats.
// ---------------------------------------------------------------------------
__global__ void k_convert(const float* __restrict__ hs,
                          const float* __restrict__ Wq, const float* __restrict__ Wk,
                          const float* __restrict__ Wv, const float* __restrict__ Wo,
                          _Float16* __restrict__ Xh,
                          _Float16* __restrict__ Wqh, _Float16* __restrict__ Wkh,
                          _Float16* __restrict__ Wvh, _Float16* __restrict__ Woh) {
    const long NX = (long)B_ * SD_ * HID_;
    const long NW = (long)HID_ * HID_;
    const long total = NX + 4 * NW;
    for (long i = blockIdx.x * (long)blockDim.x + threadIdx.x; i < total;
         i += (long)gridDim.x * blockDim.x) {
        if (i < NX) {
            long col = i % HID_;
            long row = i / HID_;
            long b = row / SD_, qp = row % SD_;
            Xh[i] = (_Float16)hs[((size_t)b * S_ + 2 * qp) * HID_ + col];
        } else {
            long j = i - NX;
            int  w = (int)(j / NW);
            long e = j % NW;
            const float* src = (w == 0) ? Wq : (w == 1) ? Wk : (w == 2) ? Wv : Wo;
            _Float16*    dst = (w == 0) ? Wqh : (w == 1) ? Wkh : (w == 2) ? Wvh : Woh;
            dst[e] = (_Float16)src[e];
        }
    }
}

// ---------------------------------------------------------------------------
// Kernel 1: QKV projection.  Y = X @ W^T on even rows.
//   Q, K stored per-head (B, NH, SD, HD) f16; Q pre-scaled by 1/sqrt(HD).
//   V stored TRANSPOSED per-head (B, NH, HD, SD) f16 so attention's P@V
//   B-operands are contiguous along k.
// grid = (B*SD/16, HID/64, 3), block = 128 (4 waves, one 16-col tile each).
// ---------------------------------------------------------------------------
__global__ void k_qkv(const _Float16* __restrict__ Xh,
                      const _Float16* __restrict__ Wqh,
                      const _Float16* __restrict__ Wkh,
                      const _Float16* __restrict__ Wvh,
                      _Float16* __restrict__ Qh, _Float16* __restrict__ Kh,
                      _Float16* __restrict__ Vt) {
    const int wave = threadIdx.x >> 5;
    const int lane = threadIdx.x & 31;
    const int hi   = lane >> 4;
    const int m0   = blockIdx.x * 16;
    const int n0   = blockIdx.y * 64 + wave * 16;
    const int which = blockIdx.z;
    const _Float16* W = (which == 0) ? Wqh : (which == 1) ? Wkh : Wvh;
    const float scale = (which == 0) ? 0.125f : 1.0f;   // 1/sqrt(64) for Q

    v8f acc = {};
#pragma unroll 4
    for (int k0 = 0; k0 < HID_; k0 += 32) {
        v16h a = wmma_load_contig(Xh, HID_, m0, k0);
        v16h b = wmma_load_contig(W,  HID_, n0, k0);
        acc = wmma_f16(a, b, acc);
    }
    const int colg = n0 + (lane & 15);
    const int h = colg >> 6, d = colg & 63;
    if (which == 2) {           // V: transposed per-head store (d-major)
#pragma unroll
        for (int r = 0; r < 8; ++r) {
            int mg = m0 + r + hi * 8;
            int b  = mg / SD_, qp = mg % SD_;
            Vt[(((size_t)b * NH_ + h) * HD_ + d) * SD_ + qp] = (_Float16)acc[r];
        }
    } else {                    // Q / K: row-major per-head store
        _Float16* Y = (which == 0) ? Qh : Kh;
#pragma unroll
        for (int r = 0; r < 8; ++r) {
            int mg = m0 + r + hi * 8;
            int b  = mg / SD_, qp = mg % SD_;
            Y[(((size_t)b * NH_ + h) * SD_ + qp) * HD_ + d] = (_Float16)(acc[r] * scale);
        }
    }
}

// ---------------------------------------------------------------------------
// Kernel 2: causal flash attention, one wave per 16-row q-tile.
// grid = (SD/16/4, NH, B), block = 128 (4 independent waves -> no __syncthreads
// inside the causally-divergent loop; per-wave s_wait_dscnt instead).
// Output Ah: (B, SD, NH*HD) f16, ready to be the A operand of the out-proj.
// ---------------------------------------------------------------------------
#define QT_PER_BLOCK 4
__global__ void k_attn(const _Float16* __restrict__ Qh,
                       const _Float16* __restrict__ Kh,
                       const _Float16* __restrict__ Vt,
                       _Float16* __restrict__ Ah) {
    __shared__ _Float16 pstage[QT_PER_BLOCK][16 * 32];
    const int wave = threadIdx.x >> 5;
    const int lane = threadIdx.x & 31;
    const int hi   = lane >> 4;
    const int colL = lane & 15;
    const int h = blockIdx.y, b = blockIdx.z;
    const int qt  = blockIdx.x * QT_PER_BLOCK + wave;
    const int qr0 = qt * 16;

    const size_t head = ((size_t)b * NH_ + h) * SD_ * HD_;
    const _Float16* Qp  = Qh + head;
    const _Float16* Kp  = Kh + head;
    const _Float16* Vtp = Vt + head;      // per-head Vt: (HD, SD), d-major
    _Float16* ps = pstage[wave];

    // Q A-operands for the whole tile (d = 0..31, 32..63), loaded once.
    v16h aq0 = wmma_load_contig(Qp, HD_, qr0, 0);
    v16h aq1 = wmma_load_contig(Qp, HD_, qr0, 32);

    float m[8], l[8];
    v8f o[4] = {{}, {}, {}, {}};
#pragma unroll
    for (int r = 0; r < 8; ++r) { m[r] = NEG_INF; l[r] = 0.0f; }

    const int nch = (qr0 + 16 + 31) >> 5;     // 32-wide k chunks, causal bound
    for (int c = 0; c < nch; ++c) {
        const int kc0 = c * 32;

        // Prefetch next chunk of K (contiguous 4KB) and Vt (64 row segments)
        // into the cache hierarchy while this chunk's softmax VALU runs.
        if (c + 1 < nch) {
            const int kn = kc0 + 32;
            __builtin_prefetch(Kp + (size_t)kn * HD_ + lane * 128, 0, 3);
            __builtin_prefetch(Vtp + (size_t)lane        * SD_ + kn, 0, 3);
            __builtin_prefetch(Vtp + (size_t)(lane + 32) * SD_ + kn, 0, 3);
        }

        // S = Q K^T for a 16x32 chunk: 4 wmma (two 16-col subtiles, d split 2x).
        v8f s0 = {}, s1 = {};
        s0 = wmma_f16(aq0, wmma_load_contig(Kp, HD_, kc0,      0),  s0);
        s0 = wmma_f16(aq1, wmma_load_contig(Kp, HD_, kc0,      32), s0);
        s1 = wmma_f16(aq0, wmma_load_contig(Kp, HD_, kc0 + 16, 0),  s1);
        s1 = wmma_f16(aq1, wmma_load_contig(Kp, HD_, kc0 + 16, 32), s1);

        // Causal mask (only bites on the diagonal chunk).
        const int colg0 = kc0 + colL;
#pragma unroll
        for (int r = 0; r < 8; ++r) {
            int rowg = qr0 + r + hi * 8;
            if (colg0      > rowg) s0[r] = NEG_INF;
            if (colg0 + 16 > rowg) s1[r] = NEG_INF;
        }

        // Online softmax update; stage P (f16) in LDS in (row, col) order.
        float sc[8];
#pragma unroll
        for (int r = 0; r < 8; ++r) {
            float mx = fmaxf(rowmax16(fmaxf(s0[r], s1[r])), m[r]);
            float p0 = __expf(s0[r] - mx);
            float p1 = __expf(s1[r] - mx);
            float scale = __expf(m[r] - mx);
            int row = r + hi * 8;
            ps[row * 32 + colL]      = (_Float16)p0;
            ps[row * 32 + 16 + colL] = (_Float16)p1;
            l[r] = l[r] * scale + rowsum16(p0 + p1);
            m[r] = mx;
            sc[r] = scale;
        }
#pragma unroll
        for (int t = 0; t < 4; ++t)
#pragma unroll
            for (int r = 0; r < 8; ++r) o[t][r] *= sc[r];

        // Per-wave LDS fence (no block barrier: waves have different trip counts).
        asm volatile("s_wait_dscnt 0x0" ::: "memory");

        // O += P @ V : P as 16x32 A-operand; Vt rows give contiguous B tiles.
        v16h ap = wmma_load_contig(ps, 32, 0, 0);
#pragma unroll
        for (int t = 0; t < 4; ++t)
            o[t] = wmma_f16(ap, wmma_load_contig(Vtp, SD_, t * 16, kc0), o[t]);
    }

    // Epilogue: divide by row sums, store (B, SD, NH*HD) f16.
#pragma unroll
    for (int r = 0; r < 8; ++r) {
        int row = r + hi * 8;
        float invl = 1.0f / l[r];
        size_t base = ((size_t)b * SD_ + (qr0 + row)) * HID_ + h * HD_;
#pragma unroll
        for (int t = 0; t < 4; ++t)
            Ah[base + t * 16 + colL] = (_Float16)(o[t][r] * invl);
    }
}

// ---------------------------------------------------------------------------
// Kernel 3a: fill every output row with the bias (odd rows keep it; even rows
// are overwritten by the projection GEMM).
// ---------------------------------------------------------------------------
__global__ void k_fill_out(const float* __restrict__ bo, float* __restrict__ out) {
    const long total = (long)B_ * S_ * HID_;
    for (long i = blockIdx.x * (long)blockDim.x + threadIdx.x; i < total;
         i += (long)gridDim.x * blockDim.x)
        out[i] = bo[i % HID_];
}

// ---------------------------------------------------------------------------
// Kernel 3b: output projection for even rows: out[2*qp] = Ah @ Wo^T + bo.
// grid = (B*SD/16, HID/64), block = 128.
// ---------------------------------------------------------------------------
__global__ void k_oproj(const _Float16* __restrict__ Ah,
                        const _Float16* __restrict__ Woh,
                        const float* __restrict__ bo, float* __restrict__ out) {
    const int wave = threadIdx.x >> 5;
    const int lane = threadIdx.x & 31;
    const int hi   = lane >> 4;
    const int m0   = blockIdx.x * 16;
    const int n0   = blockIdx.y * 64 + wave * 16;

    v8f acc = {};
#pragma unroll 4
    for (int k0 = 0; k0 < HID_; k0 += 32)
        acc = wmma_f16(wmma_load_contig(Ah, HID_, m0, k0),
                       wmma_load_contig(Woh, HID_, n0, k0), acc);

    const int colg = n0 + (lane & 15);
    const float bias = bo[colg];
#pragma unroll
    for (int r = 0; r < 8; ++r) {
        int mg = m0 + r + hi * 8;
        int b  = mg / SD_, qp = mg % SD_;
        out[((size_t)b * S_ + 2 * qp) * HID_ + colg] = acc[r] + bias;
    }
}

// ---------------------------------------------------------------------------
extern "C" void kernel_launch(void* const* d_in, const int* in_sizes, int n_in,
                              void* d_out, int out_size, void* d_ws, size_t ws_size,
                              hipStream_t stream) {
    const float* hs = (const float*)d_in[0];
    const float* Wq = (const float*)d_in[1];
    const float* Wk = (const float*)d_in[2];
    const float* Wv = (const float*)d_in[3];
    const float* Wo = (const float*)d_in[4];
    const float* bo = (const float*)d_in[5];
    float* out = (float*)d_out;

    // Workspace layout (f16), ~48 MiB total.
    char* ws = (char*)d_ws;
    size_t off = 0;
    const size_t XBYTES = (size_t)B_ * SD_ * HID_ * sizeof(_Float16);  // 8 MiB
    const size_t WBYTES = (size_t)HID_ * HID_ * sizeof(_Float16);      // 2 MiB
    const size_t HBYTES = (size_t)B_ * NH_ * SD_ * HD_ * sizeof(_Float16); // 8 MiB
    _Float16* Xh  = (_Float16*)(ws + off); off += XBYTES;
    _Float16* Wqh = (_Float16*)(ws + off); off += WBYTES;
    _Float16* Wkh = (_Float16*)(ws + off); off += WBYTES;
    _Float16* Wvh = (_Float16*)(ws + off); off += WBYTES;
    _Float16* Woh = (_Float16*)(ws + off); off += WBYTES;
    _Float16* Qh  = (_Float16*)(ws + off); off += HBYTES;
    _Float16* Kh  = (_Float16*)(ws + off); off += HBYTES;
    _Float16* Vt  = (_Float16*)(ws + off); off += HBYTES;
    _Float16* Ah  = (_Float16*)(ws + off); off += XBYTES;

    k_convert<<<1024, 256, 0, stream>>>(hs, Wq, Wk, Wv, Wo,
                                        Xh, Wqh, Wkh, Wvh, Woh);

    k_qkv<<<dim3((B_ * SD_) / 16, HID_ / 64, 3), 128, 0, stream>>>(
        Xh, Wqh, Wkh, Wvh, Qh, Kh, Vt);

    k_attn<<<dim3(SD_ / 16 / QT_PER_BLOCK, NH_, B_), 32 * QT_PER_BLOCK, 0, stream>>>(
        Qh, Kh, Vt, Ah);

    k_fill_out<<<2048, 256, 0, stream>>>(bo, out);

    k_oproj<<<dim3((B_ * SD_) / 16, HID_ / 64), 128, 0, stream>>>(
        Ah, Woh, bo, out);
}
